// EncoderBlock_40922448396604
// MI455X (gfx1250) — matrix operand order repaired and checked by reading, not compiled, not run on hardware
//
#include <hip/hip_runtime.h>

#define BB   4
#define SS   2048
#define DD   1024
#define HH   16
#define DK   64
#define DFF  4096

typedef __attribute__((ext_vector_type(16))) __bf16 v16bf;
typedef __attribute__((ext_vector_type(8)))  float  v8f;
typedef __attribute__((ext_vector_type(4)))  unsigned int v4u;
typedef __attribute__((ext_vector_type(8)))  int v8i;
typedef __attribute__((ext_vector_type(4)))  int v4i;
typedef unsigned int   u32;
typedef unsigned short u16;
typedef unsigned long long u64;

__device__ __forceinline__ u16 f2bf(float f) {
  union { float f; u32 u; } c; c.f = f;
  u32 u = c.u;
  u32 r = u + 0x7fffu + ((u >> 16) & 1u);
  return (u16)(r >> 16);
}

__device__ __forceinline__ v8f vzero8() {
  v8f z;
#pragma unroll
  for (int i = 0; i < 8; ++i) z[i] = 0.0f;
  return z;
}

// A-fragment (16x32 bf16, MxK): VGPR i holds K-pair (i&3)*2 + (i>>2)*16 + 8*half
__device__ __forceinline__ v16bf load_frag_a(const u16* p, int idx, int stride,
                                             int kbase, int lane) {
  const int half = (lane >> 4) & 1;
  union { v16bf v; u32 u[8]; } r;
#pragma unroll
  for (int i = 0; i < 8; ++i) {
    const int kp = ((i & 3) << 1) + ((i >> 2) << 4) + (half << 3) + kbase;
    r.u[i] = *(const u32*)(p + (size_t)idx * stride + kp);
  }
  return r.v;
}

// B-fragment (32x16 bf16, KxN): VGPR i holds K-pair (i&3)*2 + (i>>2)*16 + 8*half
__device__ __forceinline__ v16bf load_frag_b(const u16* p, int idx, int stride,
                                             int kbase, int lane) {
  const int half = (lane >> 4) & 1;
  union { v16bf v; u32 u[8]; } r;
#pragma unroll
  for (int i = 0; i < 8; ++i) {
    const int kp = ((i & 3) << 1) + ((i >> 2) << 4) + (half << 3) + kbase;
    r.u[i] = *(const u32*)(p + (size_t)idx * stride + kp);
  }
  return r.v;
}

// ---------------------------------------------------------------------------
// TDM: async 2-D tile (rows x 32 bf16) Global -> LDS with built-in padding
// that produces a row stride of 34 bf16 (pad_interval=16 DW, pad_amount=1 DW).
// Descriptor per CDNA5 ISA. Issued once per wave.
// ---------------------------------------------------------------------------
__device__ __forceinline__ void tdm_load_tile(u32 lds_off, const u16* gaddr,
                                              u32 tdim0, u32 tdim1,
                                              u32 tile0, u32 tile1, u32 stride0) {
  const u64 ga = (u64)(uintptr_t)gaddr;
  v4u g0;
  g0.x = 1u;                                   // count=1, gather off
  g0.y = lds_off;                              // LDS byte address
  g0.z = (u32)ga;                              // global_addr[31:0]
  g0.w = (u32)(ga >> 32) | (2u << 30);         // global_addr[56:32] | type=2
  v8i g1;
  g1[0] = (int)((1u << 16) | (1u << 20) | (3u << 22));  // 2B elems, pad_en, 16DW interval, 1DW pad
  g1[1] = (int)((tdim0 & 0xffffu) << 16);               // tensor_dim0 lo16
  g1[2] = (int)((tdim0 >> 16) | ((tdim1 & 0xffffu) << 16));
  g1[3] = (int)((tdim1 >> 16) | (tile0 << 16));         // tensor_dim1 hi | tile_dim0
  g1[4] = (int)(tile1 & 0xffffu);                       // tile_dim1 (tile_dim2=0)
  g1[5] = (int)stride0;                                 // tensor_dim0_stride[31:0]
  g1[6] = 0;
  g1[7] = 0;
  v4i g2 = {0, 0, 0, 0};
  v4i g3 = {0, 0, 0, 0};
#if __clang_major__ >= 23
  v8i g4 = {0, 0, 0, 0, 0, 0, 0, 0};
  __builtin_amdgcn_tensor_load_to_lds(g0, g1, g2, g3, g4, 0);
#else
  __builtin_amdgcn_tensor_load_to_lds(g0, g1, g2, g3, 0);
#endif
}

// ---------------------------------------------------------------------------
// fp32 -> bf16 transposed convert: out[n*K + k] = bf16(in[k*N + n])
// (weights stored [N,K] so TDM can stage B tiles without a transpose)
// ---------------------------------------------------------------------------
__global__ void cvt_bf16_t_kernel(const float* __restrict__ in, u16* __restrict__ out,
                                  int K, int N) {
  const int idx = blockIdx.x * blockDim.x + threadIdx.x;
  if (idx < K * N) {
    const int n = idx / K, k = idx - n * K;
    out[idx] = f2bf(in[(size_t)k * N + n]);
  }
}

// ---------------------------------------------------------------------------
// LayerNorm (ddof=1, alpha*(x-mean)/(std+eps)+beta) -> bf16 row
// ---------------------------------------------------------------------------
__global__ __launch_bounds__(256) void ln_bf16_kernel(
    const float* __restrict__ x, const float* __restrict__ alpha,
    const float* __restrict__ beta, u16* __restrict__ out) {
  __shared__ float red[16];
  const int row = blockIdx.x;
  const int tid = threadIdx.x;
  const float* xr = x + (size_t)row * DD;
  float v[4];
  float s1 = 0.f, s2 = 0.f;
#pragma unroll
  for (int i = 0; i < 4; ++i) {
    v[i] = xr[tid + i * 256];
    s1 += v[i];
    s2 += v[i] * v[i];
  }
#pragma unroll
  for (int m = 1; m < 32; m <<= 1) {
    s1 += __shfl_xor(s1, m);
    s2 += __shfl_xor(s2, m);
  }
  const int wave = tid >> 5;
  if ((tid & 31) == 0) { red[wave] = s1; red[8 + wave] = s2; }
  __syncthreads();
  float a1 = 0.f, a2 = 0.f;
#pragma unroll
  for (int i = 0; i < 8; ++i) { a1 += red[i]; a2 += red[8 + i]; }
  const float mean = a1 * (1.0f / DD);
  const float var  = fmaxf(a2 - (float)DD * mean * mean, 0.0f) * (1.0f / (DD - 1));
  const float inv  = 1.0f / (sqrtf(var) + 1e-6f);
#pragma unroll
  for (int i = 0; i < 4; ++i) {
    const int c = tid + i * 256;
    out[(size_t)row * DD + c] = f2bf(alpha[c] * (v[i] - mean) * inv + beta[c]);
  }
}

// ---------------------------------------------------------------------------
// bf16 GEMM: C[M,N] = A[M,K] @ Bt[N,K]^T + bias, f32 accumulate via WMMA.
// 256 threads (8 waves), tile 128(M) x 64(N), K-step 32.
// Tiles staged by the Tensor Data Mover (double buffered, TENSORcnt).
// mode 0: f32 out (+residual)   mode 1: bf16 head-split [B,H,S,DK]
// mode 2: bf16 transposed head-split [B,H,DK,S]   mode 3: bf16 + ReLU
// ---------------------------------------------------------------------------
#define ASTR 34
#define BSTR 34

__global__ __launch_bounds__(256) void gemm_bf16_kernel(
    const u16* __restrict__ A, const u16* __restrict__ Bt,
    const float* __restrict__ bias, const float* __restrict__ residual,
    void* __restrict__ outp, int M, int N, int K, int mode) {
  __shared__ u16 a_lds[2][128 * ASTR] __attribute__((aligned(16)));
  __shared__ u16 b_lds[2][64 * BSTR] __attribute__((aligned(16)));
  const int tid  = threadIdx.x;
  const int lane = tid & 31;
  const int wave = tid >> 5;
  const int l15  = lane & 15;
  const int m0   = blockIdx.y * 128;
  const int n0   = blockIdx.x * 64;

  const u32 offA[2] = { (u32)(uintptr_t)&a_lds[0][0], (u32)(uintptr_t)&a_lds[1][0] };
  const u32 offB[2] = { (u32)(uintptr_t)&b_lds[0][0], (u32)(uintptr_t)&b_lds[1][0] };

  v8f acc[4];
#pragma unroll
  for (int t = 0; t < 4; ++t) acc[t] = vzero8();

  if (wave == 0) {  // prologue: stage first K-slab into buffer 0
    tdm_load_tile(offA[0], A + (size_t)m0 * K, (u32)K, (u32)M, 32u, 128u, (u32)K);
    tdm_load_tile(offB[0], Bt + (size_t)n0 * K, (u32)K, (u32)N, 32u, 64u, (u32)K);
  }

  for (int kk = 0; kk < K; kk += 32) {
    const int buf = (kk >> 5) & 1;
    if (wave == 0) {
      if (kk + 32 < K) {  // stage next slab into the other buffer, keep it in flight
        tdm_load_tile(offA[buf ^ 1], A + (size_t)m0 * K + kk + 32, (u32)K, (u32)M, 32u, 128u, (u32)K);
        tdm_load_tile(offB[buf ^ 1], Bt + (size_t)n0 * K + kk + 32, (u32)K, (u32)N, 32u, 64u, (u32)K);
        __builtin_amdgcn_s_wait_tensorcnt(2);  // current slab's two loads done
      } else {
        __builtin_amdgcn_s_wait_tensorcnt(0);
      }
    }
    __syncthreads();
    const v16bf af = load_frag_a(a_lds[buf], wave * 16 + l15, ASTR, 0, lane);
#pragma unroll
    for (int t = 0; t < 4; ++t) {
      const v16bf bfr = load_frag_a(b_lds[buf], t * 16 + l15, BSTR, 0, lane);
      // b_lds holds B^T rows = columns of B; A-style LDS read order, but the
      // operand fed to WMMA must use the B-fragment K mapping:
      (void)bfr;
      const v16bf bop = load_frag_b(b_lds[buf], t * 16 + l15, BSTR, 0, lane);
      acc[t] = __builtin_amdgcn_wmma_f32_16x16x32_bf16(
          false, af, false, bop, (short)0, acc[t], false, false);
    }
    __syncthreads();
  }

  const int half = lane >> 4;
#pragma unroll
  for (int t = 0; t < 4; ++t) {
#pragma unroll
    for (int r = 0; r < 8; ++r) {
      const int row = m0 + wave * 16 + r + 8 * half;
      const int col = n0 + t * 16 + l15;
      float val = acc[t][r] + bias[col];
      if (mode == 0) {
        float* out = (float*)outp;
        if (residual) val += residual[(size_t)row * N + col];
        out[(size_t)row * N + col] = val;
      } else if (mode == 1) {           // q/k: [B,H,S,DK]
        u16* out = (u16*)outp;
        const int b = row >> 11, s = row & (SS - 1);
        const int h = col >> 6,  dk = col & 63;
        out[(((size_t)(b * HH + h) * SS + s) << 6) + dk] = f2bf(val);
      } else if (mode == 2) {           // v: [B,H,DK,S]
        u16* out = (u16*)outp;
        const int b = row >> 11, s = row & (SS - 1);
        const int h = col >> 6,  dk = col & 63;
        out[((size_t)(b * HH + h) * DK + dk) * SS + s] = f2bf(val);
      } else {                          // FFN1: ReLU, bf16
        u16* out = (u16*)outp;
        out[(size_t)row * N + col] = f2bf(fmaxf(val, 0.0f));
      }
    }
  }
}

// ---------------------------------------------------------------------------
// Flash attention: one wave per (b, h, 16-query tile). DK=64.
// ---------------------------------------------------------------------------
__global__ __launch_bounds__(32) void attn_kernel(
    const u16* __restrict__ q, const u16* __restrict__ k,
    const u16* __restrict__ vt, const int* __restrict__ mask,
    u16* __restrict__ ctx) {
  __shared__ u16 lds_p[16 * 32] __attribute__((aligned(16)));
  const int lane = threadIdx.x;
  const int l15  = lane & 15;
  const int half = lane >> 4;
  const int q0 = blockIdx.x * 16;
  const int h  = blockIdx.y;
  const int b  = blockIdx.z;
  const int bh = b * HH + h;
  const u16* qp = q  + ((size_t)bh * SS + q0) * DK;
  const u16* kp = k  + (size_t)bh * SS * DK;
  const u16* vp = vt + (size_t)bh * DK * SS;
  const int* mp = mask + (size_t)b * SS;

  const v16bf qa0 = load_frag_a(qp, l15, DK, 0,  lane);
  const v16bf qa1 = load_frag_a(qp, l15, DK, 32, lane);

  const float NEG = -__builtin_inff();
  float m_r[8], l_r[8];
  v8f o[4];
#pragma unroll
  for (int r = 0; r < 8; ++r) { m_r[r] = NEG; l_r[r] = 0.0f; }
#pragma unroll
  for (int t = 0; t < 4; ++t) o[t] = vzero8();

  const float scale = 0.125f;  // 1/sqrt(DK)

  for (int j = 0; j < SS; j += 32) {
    if (j + 32 < SS) {  // stream-ahead hint for next K tile (global_prefetch_b8)
      __builtin_prefetch(kp + (size_t)(j + 32 + l15) * DK, 0, 0);
    }
    const v16bf kb00 = load_frag_b(kp, j + l15,      DK, 0,  lane);
    const v16bf kb01 = load_frag_b(kp, j + l15,      DK, 32, lane);
    const v16bf kb10 = load_frag_b(kp, j + 16 + l15, DK, 0,  lane);
    const v16bf kb11 = load_frag_b(kp, j + 16 + l15, DK, 32, lane);
    v8f s0 = __builtin_amdgcn_wmma_f32_16x16x32_bf16(
        false, qa0, false, kb00, (short)0, vzero8(), false, false);
    s0 = __builtin_amdgcn_wmma_f32_16x16x32_bf16(
        false, qa1, false, kb01, (short)0, s0, false, false);
    v8f s1 = __builtin_amdgcn_wmma_f32_16x16x32_bf16(
        false, qa0, false, kb10, (short)0, vzero8(), false, false);
    s1 = __builtin_amdgcn_wmma_f32_16x16x32_bf16(
        false, qa1, false, kb11, (short)0, s1, false, false);

    const int mv0 = mp[j + l15];
    const int mv1 = mp[j + 16 + l15];

#pragma unroll
    for (int r = 0; r < 8; ++r) {
      const float x0 = mv0 ? s0[r] * scale : NEG;
      const float x1 = mv1 ? s1[r] * scale : NEG;
      float red = fmaxf(x0, x1);
      red = fmaxf(red, __shfl_xor(red, 1));
      red = fmaxf(red, __shfl_xor(red, 2));
      red = fmaxf(red, __shfl_xor(red, 4));
      red = fmaxf(red, __shfl_xor(red, 8));
      const float mn = fmaxf(m_r[r], red);
      const float f  = __expf(m_r[r] - mn);
      const float p0 = __expf(x0 - mn);
      const float p1 = __expf(x1 - mn);
      float ps = p0 + p1;
      ps += __shfl_xor(ps, 1);
      ps += __shfl_xor(ps, 2);
      ps += __shfl_xor(ps, 4);
      ps += __shfl_xor(ps, 8);
      l_r[r] = l_r[r] * f + ps;
      m_r[r] = mn;
#pragma unroll
      for (int t = 0; t < 4; ++t) o[t][r] *= f;
      lds_p[(r + 8 * half) * 32 + l15]      = f2bf(p0);
      lds_p[(r + 8 * half) * 32 + 16 + l15] = f2bf(p1);
    }
    asm volatile("s_wait_dscnt 0" ::: "memory");
    const v16bf pa = load_frag_a(lds_p, l15, 32, 0, lane);
#pragma unroll
    for (int t = 0; t < 4; ++t) {
      const v16bf vb = load_frag_b(vp, t * 16 + l15, SS, j, lane);
      o[t] = __builtin_amdgcn_wmma_f32_16x16x32_bf16(
          false, pa, false, vb, (short)0, o[t], false, false);
    }
  }

#pragma unroll
  for (int t = 0; t < 4; ++t) {
#pragma unroll
    for (int r = 0; r < 8; ++r) {
      const int row = q0 + r + 8 * half;
      const int col = h * DK + t * 16 + l15;
      ctx[(size_t)(b * SS + row) * DD + col] = f2bf(o[t][r] / l_r[r]);
    }
  }
}

// ---------------------------------------------------------------------------
extern "C" void kernel_launch(void* const* d_in, const int* in_sizes, int n_in,
                              void* d_out, int out_size, void* d_ws, size_t ws_size,
                              hipStream_t stream) {
  (void)in_sizes; (void)n_in; (void)out_size; (void)ws_size;
  const float* x    = (const float*)d_in[0];
  const int*   mask = (const int*)d_in[1];
  const float* wq   = (const float*)d_in[2];
  const float* bq   = (const float*)d_in[3];
  const float* wk   = (const float*)d_in[4];
  const float* bk   = (const float*)d_in[5];
  const float* wv   = (const float*)d_in[6];
  const float* bv   = (const float*)d_in[7];
  const float* wo   = (const float*)d_in[8];
  const float* bo   = (const float*)d_in[9];
  const float* ln1a = (const float*)d_in[10];
  const float* ln1b = (const float*)d_in[11];
  const float* ln2a = (const float*)d_in[12];
  const float* ln2b = (const float*)d_in[13];
  const float* w1   = (const float*)d_in[14];
  const float* b1   = (const float*)d_in[15];
  const float* w2   = (const float*)d_in[16];
  const float* b2   = (const float*)d_in[17];

  const int MT = BB * SS;  // 8192 rows

  char*  ws  = (char*)d_ws;
  size_t off = 0;
  auto alloc = [&](size_t bytes) -> void* {
    void* p = ws + off;
    off = (off + bytes + 255) & ~(size_t)255;
    return p;
  };
  u16*   xn   = (u16*)alloc((size_t)MT * DD * 2);   // reused for xn2
  u16*   wqt  = (u16*)alloc((size_t)DD * DD * 2);
  u16*   wkt  = (u16*)alloc((size_t)DD * DD * 2);
  u16*   wvt  = (u16*)alloc((size_t)DD * DD * 2);
  u16*   wot  = (u16*)alloc((size_t)DD * DD * 2);
  u16*   w1t  = (u16*)alloc((size_t)DD * DFF * 2);
  u16*   w2t  = (u16*)alloc((size_t)DFF * DD * 2);
  u16*   qb   = (u16*)alloc((size_t)MT * DD * 2);
  u16*   kbuf = (u16*)alloc((size_t)MT * DD * 2);
  u16*   vtb  = (u16*)alloc((size_t)MT * DD * 2);
  u16*   ctxb = (u16*)alloc((size_t)MT * DD * 2);
  float* x2   = (float*)alloc((size_t)MT * DD * 4);
  u16*   hb   = (u16*)alloc((size_t)MT * DFF * 2);

  const int nw  = DD * DD;
  const int nw1 = DD * DFF;
  cvt_bf16_t_kernel<<<(nw  + 255) / 256, 256, 0, stream>>>(wq, wqt, DD, DD);
  cvt_bf16_t_kernel<<<(nw  + 255) / 256, 256, 0, stream>>>(wk, wkt, DD, DD);
  cvt_bf16_t_kernel<<<(nw  + 255) / 256, 256, 0, stream>>>(wv, wvt, DD, DD);
  cvt_bf16_t_kernel<<<(nw  + 255) / 256, 256, 0, stream>>>(wo, wot, DD, DD);
  cvt_bf16_t_kernel<<<(nw1 + 255) / 256, 256, 0, stream>>>(w1, w1t, DD, DFF);
  cvt_bf16_t_kernel<<<(nw1 + 255) / 256, 256, 0, stream>>>(w2, w2t, DFF, DD);

  ln_bf16_kernel<<<MT, 256, 0, stream>>>(x, ln1a, ln1b, xn);

  const dim3 gD(DD / 64, MT / 128);
  gemm_bf16_kernel<<<gD, 256, 0, stream>>>(xn, wqt, bq, nullptr, qb,   MT, DD, DD, 1);
  gemm_bf16_kernel<<<gD, 256, 0, stream>>>(xn, wkt, bk, nullptr, kbuf, MT, DD, DD, 1);
  gemm_bf16_kernel<<<gD, 256, 0, stream>>>(xn, wvt, bv, nullptr, vtb,  MT, DD, DD, 2);

  attn_kernel<<<dim3(SS / 16, HH, BB), 32, 0, stream>>>(qb, kbuf, vtb, mask, ctxb);

  gemm_bf16_kernel<<<gD, 256, 0, stream>>>(ctxb, wot, bo, x, x2, MT, DD, DD, 0);

  ln_bf16_kernel<<<MT, 256, 0, stream>>>(x2, ln2a, ln2b, xn);

  gemm_bf16_kernel<<<dim3(DFF / 64, MT / 128), 256, 0, stream>>>(
      xn, w1t, b1, nullptr, hb, MT, DFF, DD, 3);

  gemm_bf16_kernel<<<gD, 256, 0, stream>>>(hb, w2t, b2, x2, d_out, MT, DD, DFF, 0);
}
